// UTransNet_90117003804770
// MI455X (gfx1250) — matrix operand order repaired and checked by reading, not compiled, not run on hardware
//
#include <hip/hip_runtime.h>
#include <math.h>

// ---------------------------------------------------------------------------
// UTransNet forward for MI455X (gfx1250, wave32, WMMA).
//
// Roofline: ~10 GFLOP total, ~150 MB live data (fits in 192 MB L2, HBM floor
// ~5us at 23.3 TB/s). Critical path = 2000-step serial Mamba recurrence, so
// the scan kernel is kept minimal (16 lanes per (b,d) state, shfl_xor
// reduction, fused SiLU gate). All big projections use
// v_wmma_f32_16x16x32_f16 (f16 operands, f32 accumulation), K-unrolled x2
// with dual accumulators so consecutive WMMAs have no D->C dependency and
// the XDL pipe stays busy. Attention has head_dim=8 (75% WMMA padding) and
// is L2-resident -> VALU flash-attention with LDS-staged K/V tiles.
// ---------------------------------------------------------------------------

typedef __attribute__((ext_vector_type(16))) _Float16 v16h;
typedef __attribute__((ext_vector_type(8)))  float    v8f;

#define Bsz 16
#define Lsz 2000
#define Dm  64
#define DI  128
#define DS  16
#define DR  4
#define L2s 1000   // after branch maxpool

// ---------------------------------------------------------------------------
// Generic WMMA GEMM: C[M,N] = act(A[M,K] @ W[N,K]^T + bias + residual)
// Requirements: M % 16 == 0, K % 64 == 0 (true for all uses: K in {64,128,32000}).
// 128-thread blocks = 4 waves; each wave owns one 16x16 output tile, waves
// cover 4 adjacent N-tiles (A rows shared via L0/L2). K unrolled by 64 with
// two independent f32 accumulators -> back-to-back v_wmma issue.
// ---------------------------------------------------------------------------
__global__ __launch_bounds__(128)
void gemm_wmma_kernel(const float* __restrict__ A,
                      const float* __restrict__ W,
                      const float* __restrict__ bias,
                      const float* __restrict__ residual,
                      float* __restrict__ C,
                      int M, int N, int K, int act)
{
    const int lane = threadIdx.x & 31;     // lane in wave32
    const int wid  = threadIdx.x >> 5;     // wave id in block (0..3)
    const int col  = lane & 15;
    const int hi   = lane >> 4;            // 0: lanes 0-15, 1: lanes 16-31
    const int n0   = (blockIdx.x * 4 + wid) * 16;
    const int m0   = blockIdx.y * 16;

    // C/D layout: VGPR r, lanes 0-15 -> (M=r, N=lane); lanes 16-31 -> (M=8+r).
    v8f acc0, acc1;
    float binit = 0.0f;
    if (bias != nullptr && (n0 + col) < N) binit = bias[n0 + col];
#pragma unroll
    for (int r = 0; r < 8; ++r) { acc0[r] = binit; acc1[r] = 0.0f; }

    const int  m   = m0 + col;             // A row for this lane (both halves)
    const int  n   = n0 + col;             // W row (output column)
    const bool nok = (n < N);

    for (int kk = 0; kk < K; kk += 64) {
        // A fragment, 16x32 f16 layout:
        //  lanes 0-15 : elems 0..7 = K k0+0..7,  elems 8..15 = K k0+16..23
        //  lanes 16-31: elems 0..7 = K k0+8..15, elems 8..15 = K k0+24..31
        v16h af0, af1, bf0, bf1;
        const float* arow = A + (size_t)m * K + kk + hi * 8;
#pragma unroll
        for (int e = 0; e < 8; ++e) af0[e]     = (_Float16)arow[e];
#pragma unroll
        for (int e = 0; e < 8; ++e) af0[8 + e] = (_Float16)arow[16 + e];
#pragma unroll
        for (int e = 0; e < 8; ++e) af1[e]     = (_Float16)arow[32 + e];
#pragma unroll
        for (int e = 0; e < 8; ++e) af1[8 + e] = (_Float16)arow[48 + e];

        // B fragment, same K order as A: lanes 0-15 hold K k0+0..7,16..23 for col n,
        // lanes 16-31 hold K k0+8..15,24..31.  B[k][n] = W[n][k].
        if (nok) {
            const float* wrow = W + (size_t)n * K + kk + hi * 8;
#pragma unroll
            for (int e = 0; e < 16; ++e) bf0[e] = (_Float16)wrow[e + (e >> 3) * 8];
#pragma unroll
            for (int e = 0; e < 16; ++e) bf1[e] = (_Float16)wrow[32 + e + (e >> 3) * 8];
        } else {
#pragma unroll
            for (int e = 0; e < 16; ++e) { bf0[e] = (_Float16)0.0f; bf1[e] = (_Float16)0.0f; }
        }

        acc0 = __builtin_amdgcn_wmma_f32_16x16x32_f16(
                   false, af0, false, bf0, (short)0, acc0, false, false);
        acc1 = __builtin_amdgcn_wmma_f32_16x16x32_f16(
                   false, af1, false, bf1, (short)0, acc1, false, false);
    }

    if (nok) {
#pragma unroll
        for (int r = 0; r < 8; ++r) {
            int row = m0 + r + hi * 8;
            float v = acc0[r] + acc1[r];
            if (residual != nullptr) v += residual[(size_t)row * N + n];
            if (act == 1) v = fmaxf(v, 0.0f);
            C[(size_t)row * N + n] = v;
        }
    }
}

// ---------------------------------------------------------------------------
// Branch front end: conv1d(1->64, k=3, pad 1) + ELU + LayerNorm(64) fused.
// One thread per (b,l) position; conv recomputed in 2nd pass to avoid a
// 64-register spill array (3 MACs per channel, trivially cheap).
// ---------------------------------------------------------------------------
__global__ void conv3_elu_ln_kernel(const float* __restrict__ seq,
                                    const float* __restrict__ w3,
                                    const float* __restrict__ b3,
                                    const float* __restrict__ g,
                                    const float* __restrict__ bln,
                                    float* __restrict__ xl)
{
    int idx = blockIdx.x * blockDim.x + threadIdx.x;   // b*L + l
    if (idx >= Bsz * Lsz) return;
    int b = idx / Lsz, l = idx % Lsz;
    float sm1 = (l > 0)       ? seq[b * Lsz + l - 1] : 0.0f;
    float s0  =                 seq[b * Lsz + l];
    float sp1 = (l < Lsz - 1) ? seq[b * Lsz + l + 1] : 0.0f;

    float mu = 0.0f, m2 = 0.0f;
    for (int c = 0; c < Dm; ++c) {
        float v = w3[c*3+0]*sm1 + w3[c*3+1]*s0 + w3[c*3+2]*sp1 + b3[c];
        v = (v > 0.0f) ? v : (expf(v) - 1.0f);         // ELU(alpha=1)
        mu += v; m2 += v * v;
    }
    mu *= (1.0f / Dm);
    float var = m2 * (1.0f / Dm) - mu * mu;
    float inv = rsqrtf(var + 1e-5f);
    for (int c = 0; c < Dm; ++c) {
        float v = w3[c*3+0]*sm1 + w3[c*3+1]*s0 + w3[c*3+2]*sp1 + b3[c];
        v = (v > 0.0f) ? v : (expf(v) - 1.0f);
        xl[(size_t)idx * Dm + c] = (v - mu) * inv * g[c] + bln[c];
    }
}

// ---------------------------------------------------------------------------
// Depthwise causal conv (k=4, pad(3,0)) over xz[...,:128] + SiLU -> xm.
// ---------------------------------------------------------------------------
__global__ void dwconv_silu_kernel(const float* __restrict__ xz,
                                   const float* __restrict__ cw,
                                   const float* __restrict__ cb,
                                   float* __restrict__ xm)
{
    int idx = blockIdx.x * blockDim.x + threadIdx.x;   // (b*L+l)*128 + d
    if (idx >= Bsz * Lsz * DI) return;
    int d  = idx & (DI - 1);
    int bl = idx / DI;
    int l  = bl % Lsz, b = bl / Lsz;
    float acc = cb[d];
#pragma unroll
    for (int j = 0; j < 4; ++j) {
        int li = l - 3 + j;
        if (li >= 0)
            acc += cw[d*4 + j] * xz[((size_t)(b * Lsz + li)) * 256 + d];
    }
    float sg = 1.0f / (1.0f + expf(-acc));
    xm[(size_t)bl * DI + d] = acc * sg;                // SiLU
}

// ---------------------------------------------------------------------------
// delta = softplus(dt @ m_dt_w^T + m_dt_b), dt = xdb[...,:4]
// ---------------------------------------------------------------------------
__global__ void delta_kernel(const float* __restrict__ xdb,
                             const float* __restrict__ dtw,
                             const float* __restrict__ dtb,
                             float* __restrict__ delta)
{
    int idx = blockIdx.x * blockDim.x + threadIdx.x;
    if (idx >= Bsz * Lsz * DI) return;
    int d  = idx & (DI - 1);
    int bl = idx / DI;
    const float* row = xdb + (size_t)bl * 36;
    float v = dtb[d];
#pragma unroll
    for (int r = 0; r < DR; ++r) v += dtw[d*DR + r] * row[r];
    delta[(size_t)bl * DI + d] = (v > 20.0f) ? v : log1pf(expf(v));
}

// ---------------------------------------------------------------------------
// Mamba selective scan (the serial critical path). 16 lanes per (b,d) pair,
// one state element s per lane, cross-lane einsum via shfl_xor butterfly.
// Fuses the D-skip connection and the SiLU(z) gate into the epilogue.
// ---------------------------------------------------------------------------
__global__ void scan_kernel(const float* __restrict__ xz,
                            const float* __restrict__ xm,
                            const float* __restrict__ xdb,
                            const float* __restrict__ delta,
                            const float* __restrict__ A_log,
                            const float* __restrict__ mD,
                            float* __restrict__ yf)
{
    int tid   = blockIdx.x * blockDim.x + threadIdx.x;
    int group = tid >> 4;          // (b*128 + d)
    int s     = tid & 15;
    if (group >= Bsz * DI) return;
    int b = group >> 7;
    int d = group & (DI - 1);

    float a  = -expf(A_log[d * DS + s]);
    float Dd = mD[d];
    float h  = 0.0f;
    size_t base = (size_t)b * Lsz;

    for (int l = 0; l < Lsz; ++l) {
        size_t row = base + l;
        float dl = delta[row * DI + d];
        float xv = xm[row * DI + d];
        float Bv = xdb[row * 36 + DR + s];
        float Cv = xdb[row * 36 + DR + DS + s];
        h = __expf(dl * a) * h + dl * Bv * xv;
        float y = h * Cv;
        y += __shfl_xor(y, 1, 16);
        y += __shfl_xor(y, 2, 16);
        y += __shfl_xor(y, 4, 16);
        y += __shfl_xor(y, 8, 16);
        if (s == 0) {
            float z   = xz[row * 256 + DI + d];
            float sil = z / (1.0f + expf(-z));
            yf[row * DI + d] = (y + xv * Dd) * sil;
        }
    }
}

// ---------------------------------------------------------------------------
// maxpool2 over sequence for (b, 2*Lout, 64) row-major -> (b, Lout, 64)
// ---------------------------------------------------------------------------
__global__ void maxpool_seq64_kernel(const float* __restrict__ in,
                                     float* __restrict__ out, int Lout)
{
    int idx = blockIdx.x * blockDim.x + threadIdx.x;
    if (idx >= Bsz * Lout * 64) return;
    int c  = idx & 63;
    int t  = idx >> 6;
    int l2 = t % Lout, b = t / Lout;
    size_t i0 = ((size_t)(b * 2 * Lout + 2 * l2)) * 64 + c;
    out[idx] = fmaxf(in[i0], in[i0 + 64]);
}

// ---------------------------------------------------------------------------
// Flash-attention (online softmax), head_dim = 8, seq = 1000.
// grid = (8 qpos-chunks, 8 heads, 16 batch), block = 128 threads,
// K/V staged through LDS in 64-row tiles (read from L2-resident buffers).
// ---------------------------------------------------------------------------
__global__ void attention_kernel(const float* __restrict__ Q,
                                 const float* __restrict__ Kb,
                                 const float* __restrict__ Vb,
                                 float* __restrict__ O)
{
    __shared__ float sK[64 * 8];
    __shared__ float sV[64 * 8];
    int b = blockIdx.z, h = blockIdx.y;
    int q = blockIdx.x * 128 + threadIdx.x;
    bool valid = (q < L2s);

    float qv[8];
#pragma unroll
    for (int e = 0; e < 8; ++e)
        qv[e] = valid ? Q[((size_t)(b * L2s + q)) * 64 + h * 8 + e] : 0.0f;

    float m = -3.4e38f, ssum = 0.0f, acc[8];
#pragma unroll
    for (int e = 0; e < 8; ++e) acc[e] = 0.0f;

    for (int t0 = 0; t0 < L2s; t0 += 64) {
        int cnt = min(64, L2s - t0);
        for (int i = threadIdx.x; i < 64 * 8; i += 128) {
            int kp = t0 + (i >> 3);
            int e  = i & 7;
            float kk = 0.0f, vv = 0.0f;
            if (kp < L2s) {
                size_t o = ((size_t)(b * L2s + kp)) * 64 + h * 8 + e;
                kk = Kb[o]; vv = Vb[o];
            }
            sK[i] = kk; sV[i] = vv;
        }
        __syncthreads();
        if (valid) {
            for (int j = 0; j < cnt; ++j) {
                float sc = 0.0f;
#pragma unroll
                for (int e = 0; e < 8; ++e) sc += qv[e] * sK[j*8 + e];
                sc *= 0.35355339059f;                  // 1/sqrt(8)
                float mn    = fmaxf(m, sc);
                float scale = __expf(m - mn);
                float p     = __expf(sc - mn);
                ssum = ssum * scale + p;
#pragma unroll
                for (int e = 0; e < 8; ++e) acc[e] = acc[e] * scale + p * sV[j*8 + e];
                m = mn;
            }
        }
        __syncthreads();
    }
    if (valid) {
        float inv = 1.0f / ssum;
#pragma unroll
        for (int e = 0; e < 8; ++e)
            O[((size_t)(b * L2s + q)) * 64 + h * 8 + e] = acc[e] * inv;
    }
}

// ---------------------------------------------------------------------------
// Fusion conv1d(64->64, k=3, pad 1) + ReLU on (b, l, ci) rows.
// ---------------------------------------------------------------------------
__global__ void fusion_conv_kernel(const float* __restrict__ a,
                                   const float* __restrict__ w,   // [co][ci][3]
                                   const float* __restrict__ bias,
                                   float* __restrict__ out)
{
    int idx = blockIdx.x * blockDim.x + threadIdx.x;
    if (idx >= Bsz * L2s * 64) return;
    int co = idx & 63;
    int t  = idx >> 6;
    int l  = t % L2s, b = t / L2s;
    float acc = bias[co];
#pragma unroll
    for (int j = 0; j < 3; ++j) {
        int li = l - 1 + j;
        if (li < 0 || li >= L2s) continue;
        const float* ar = a + ((size_t)(b * L2s + li)) * 64;
        const float* wr = w + co * 192 + j;
        for (int ci = 0; ci < 64; ++ci) acc += wr[ci * 3] * ar[ci];
    }
    out[((size_t)(b * L2s + l)) * 64 + co] = fmaxf(acc, 0.0f);
}

// ---------------------------------------------------------------------------
// maxpool2 + relayout (b,l,c) -> flattened (b, c*500 + l2) for fc1.
// ---------------------------------------------------------------------------
__global__ void pool_to_cmajor_kernel(const float* __restrict__ fused,
                                      float* __restrict__ pooled)
{
    int idx = blockIdx.x * blockDim.x + threadIdx.x;
    if (idx >= Bsz * 64 * 500) return;
    int l2 = idx % 500;
    int t  = idx / 500;
    int c  = t & 63, b = t >> 6;
    size_t i0 = ((size_t)(b * L2s + 2 * l2)) * 64 + c;
    pooled[(size_t)b * 32000 + c * 500 + l2] = fmaxf(fused[i0], fused[i0 + 64]);
}

// ---------------------------------------------------------------------------
// Final task heads: t1 (16x5) then t2 (16x3), concatenated into d_out.
// ---------------------------------------------------------------------------
__global__ void heads_kernel(const float* __restrict__ feat,
                             const float* __restrict__ t1w, const float* __restrict__ t1b,
                             const float* __restrict__ t2w, const float* __restrict__ t2b,
                             float* __restrict__ out)
{
    int tid = blockIdx.x * blockDim.x + threadIdx.x;
    if (tid >= Bsz * 8) return;
    int b = tid >> 3, n = tid & 7;
    const float* f = feat + b * 128;
    if (n < 5) {
        float v = t1b[n];
        for (int k = 0; k < 128; ++k) v += f[k] * t1w[n * 128 + k];
        out[b * 5 + n] = v;
    } else {
        int n2 = n - 5;
        float v = t2b[n2];
        for (int k = 0; k < 128; ++k) v += f[k] * t2w[n2 * 128 + k];
        out[Bsz * 5 + b * 3 + n2] = v;
    }
}

// ---------------------------------------------------------------------------
// Host-side orchestration.
// Input order (setup_inputs insertion order, params flattened recursively):
//  0..2 : seq1..seq3
//  3+15p: branch p: w3,b3,ln_g,ln_b,m_in,m_cw,m_cb,m_x,m_dt_w,m_dt_b,
//                   A_log,m_D,m_out,fc_w,fc_b
//  48..59: fusion_w,fusion_b,in_w,in_b,out_w,out_b,fc1_w,fc1_b,
//          t1_w,t1_b,t2_w,t2_b
// ---------------------------------------------------------------------------
extern "C" void kernel_launch(void* const* d_in, const int* in_sizes, int n_in,
                              void* d_out, int out_size, void* d_ws, size_t ws_size,
                              hipStream_t stream)
{
    (void)in_sizes; (void)n_in; (void)out_size; (void)ws_size;
    const float* seq[3] = { (const float*)d_in[0], (const float*)d_in[1],
                            (const float*)d_in[2] };

    // Workspace layout (floats). Total ~37.6M floats (~150 MB).
    float* ws = (float*)d_ws;
    const size_t BL = (size_t)Bsz * Lsz;          // 32000
    float* XL    = ws;                            // BL*64
    float* XZ    = XL    + BL * 64;               // BL*256
    float* XM    = XZ    + BL * 256;              // BL*128
    float* XDB   = XM    + BL * 128;              // BL*36
    float* DELTA = XDB   + BL * 36;               // BL*128
    float* YF    = DELTA + BL * 128;              // BL*128
    float* YMO   = YF    + BL * 128;              // BL*64
    float* FCA   = YMO   + BL * 64;               // BL*64
    float* BR0   = FCA   + BL * 64;               // 3 x 16*1000*64
    float* BR1   = BR0   + (size_t)Bsz * L2s * 64;
    float* BR2   = BR1   + (size_t)Bsz * L2s * 64;
    float* Qb    = BR2   + (size_t)Bsz * L2s * 64;
    float* Kb    = Qb    + (size_t)Bsz * L2s * 64;
    float* Vb    = Kb    + (size_t)Bsz * L2s * 64;
    float* Ob    = Vb    + (size_t)Bsz * L2s * 64;
    float* ATT   = Ob    + (size_t)Bsz * L2s * 64;
    float* FUSED = ATT   + (size_t)Bsz * L2s * 64;
    float* POOL  = FUSED + (size_t)Bsz * L2s * 64;  // 16*32000
    float* FEAT  = POOL  + (size_t)Bsz * 32000;     // 16*128
    float* BR[3] = { BR0, BR1, BR2 };

    const int TPB  = 256;
    const int M_BL = (int)BL;                     // 32000
    const int M_BQ = Bsz * L2s;                   // 16000

    for (int p = 0; p < 3; ++p) {
        void* const* bp = d_in + 3 + p * 15;
        const float* w3    = (const float*)bp[0];
        const float* b3    = (const float*)bp[1];
        const float* ln_g  = (const float*)bp[2];
        const float* ln_b  = (const float*)bp[3];
        const float* m_in  = (const float*)bp[4];
        const float* m_cw  = (const float*)bp[5];
        const float* m_cb  = (const float*)bp[6];
        const float* m_x   = (const float*)bp[7];
        const float* m_dtw = (const float*)bp[8];
        const float* m_dtb = (const float*)bp[9];
        const float* A_log = (const float*)bp[10];
        const float* m_D   = (const float*)bp[11];
        const float* m_out = (const float*)bp[12];
        const float* fc_w  = (const float*)bp[13];
        const float* fc_b  = (const float*)bp[14];

        // conv3 + ELU + LayerNorm -> XL (b,l,64)
        conv3_elu_ln_kernel<<<(M_BL + 63) / 64, 64, 0, stream>>>(
            seq[p], w3, b3, ln_g, ln_b, XL);

        // xz = XL @ m_in^T  (32000 x 256, K=64)
        gemm_wmma_kernel<<<dim3(4, M_BL / 16), 128, 0, stream>>>(
            XL, m_in, nullptr, nullptr, XZ, M_BL, 256, 64, 0);

        // depthwise causal conv + SiLU -> XM (b,l,128)
        dwconv_silu_kernel<<<(M_BL * DI + TPB - 1) / TPB, TPB, 0, stream>>>(
            XZ, m_cw, m_cb, XM);

        // xdb = XM @ m_x^T  (32000 x 36, K=128)
        gemm_wmma_kernel<<<dim3(1, M_BL / 16), 128, 0, stream>>>(
            XM, m_x, nullptr, nullptr, XDB, M_BL, 36, 128, 0);

        // delta = softplus(dt @ m_dt_w^T + b)
        delta_kernel<<<(M_BL * DI + TPB - 1) / TPB, TPB, 0, stream>>>(
            XDB, m_dtw, m_dtb, DELTA);

        // selective scan + D-skip + SiLU(z) gate -> YF (b,l,128)
        scan_kernel<<<(Bsz * DI * DS + 127) / 128, 128, 0, stream>>>(
            XZ, XM, XDB, DELTA, A_log, m_D, YF);

        // YMO = YF @ m_out^T + XL (residual)  (32000 x 64, K=128)
        gemm_wmma_kernel<<<dim3(1, M_BL / 16), 128, 0, stream>>>(
            YF, m_out, nullptr, XL, YMO, M_BL, 64, 128, 0);

        // FCA = relu(YMO @ fc_w^T + fc_b)  (32000 x 64, K=64)
        gemm_wmma_kernel<<<dim3(1, M_BL / 16), 128, 0, stream>>>(
            YMO, fc_w, fc_b, nullptr, FCA, M_BL, 64, 64, 1);

        // maxpool2 over L -> branch output (b, 1000, 64)
        maxpool_seq64_kernel<<<(Bsz * L2s * 64 + TPB - 1) / TPB, TPB, 0, stream>>>(
            FCA, BR[p], L2s);
    }

    const float* in_w  = (const float*)d_in[50];
    const float* in_b  = (const float*)d_in[51];
    const float* out_w = (const float*)d_in[52];
    const float* out_b = (const float*)d_in[53];

    // Q/K/V projections (16000 x 64, K=64) from branches 1/2/3.
    gemm_wmma_kernel<<<dim3(1, M_BQ / 16), 128, 0, stream>>>(
        BR0, in_w,            in_b,       nullptr, Qb, M_BQ, 64, 64, 0);
    gemm_wmma_kernel<<<dim3(1, M_BQ / 16), 128, 0, stream>>>(
        BR1, in_w + 64 * 64,  in_b + 64,  nullptr, Kb, M_BQ, 64, 64, 0);
    gemm_wmma_kernel<<<dim3(1, M_BQ / 16), 128, 0, stream>>>(
        BR2, in_w + 128 * 64, in_b + 128, nullptr, Vb, M_BQ, 64, 64, 0);

    // Flash attention -> Ob (b, q, h*8+hd)
    attention_kernel<<<dim3(8, 8, Bsz), 128, 0, stream>>>(Qb, Kb, Vb, Ob);

    // Output projection (16000 x 64, K=64)
    gemm_wmma_kernel<<<dim3(1, M_BQ / 16), 128, 0, stream>>>(
        Ob, out_w, out_b, nullptr, ATT, M_BQ, 64, 64, 0);

    // Fusion conv + ReLU
    fusion_conv_kernel<<<(Bsz * L2s * 64 + TPB - 1) / TPB, TPB, 0, stream>>>(
        ATT, (const float*)d_in[48], (const float*)d_in[49], FUSED);

    // maxpool2 + relayout to (b, c*500 + l2)
    pool_to_cmajor_kernel<<<(Bsz * 64 * 500 + TPB - 1) / TPB, TPB, 0, stream>>>(
        FUSED, POOL);

    // fc1: relu(POOL @ fc1_w^T + fc1_b)  (16 x 128, K=32000)
    gemm_wmma_kernel<<<dim3(2, 1), 128, 0, stream>>>(
        POOL, (const float*)d_in[54], (const float*)d_in[55], nullptr,
        FEAT, 16, 128, 32000, 1);

    // task heads -> d_out (16*5 then 16*3)
    heads_kernel<<<1, 128, 0, stream>>>(
        FEAT, (const float*)d_in[56], (const float*)d_in[57],
        (const float*)d_in[58], (const float*)d_in[59], (float*)d_out);
}
